// FlyPrompt_31602369364647
// MI455X (gfx1250) — matrix operand order repaired and checked by reading, not compiled, not run on hardware
//
#include <hip/hip_runtime.h>
#include <hip/hip_bf16.h>

// ---------------- model constants ----------------
constexpr int Bb   = 32;
constexpr int IMG  = 224;
constexpr int Dm   = 768;
constexpr int NH   = 12;
constexpr int HD   = 64;
constexpr int Lc   = 12;
constexpr int MLP  = 3072;
constexpr int S0   = 197;
constexpr int NE   = 10;
constexpr int LP   = 40;
constexpr int NPL  = 5;
constexpr int NC   = 100;
constexpr int SMAX = S0 + LP;     // 237
constexpr int SPAD = 256;         // padded seq (multiple of 32)
constexpr int TD   = 3 * Dm;      // 2304
constexpr int NP   = 196;         // 14x14 patches

typedef __attribute__((ext_vector_type(16))) _Float16 v16h;
typedef __attribute__((ext_vector_type(8)))  _Float16 v8h;
typedef __attribute__((ext_vector_type(8)))  float    v8f;

// ---------------- generic WMMA GEMM (branch-free inner loop) ----------------
// C[z][M][N] = act(alpha * A[z] @ B[z] + bias) (+ res), f16 in, f32 accum.
// z decomposed as (zb = z/nhDiv, zh = z%nhDiv) with independent strides.
// Out-of-range A rows are CLAMPED to M-1 (their outputs are never stored).
// K loop is split: main loop (k0+31 < Kvld) uses pure pointer increments;
// a short tail clamps B rows to Kvld-1 (att@v padding: softmax weights for
// k >= Kvld are exactly 0, so 0 * finite = 0).
// Block: 256 threads = 8 waves; block tile 64(M) x 64(N); each wave: 16x32.
__global__ __launch_bounds__(256)
void fp_wmma_gemm(const _Float16* __restrict__ A, int lda, long aZb, long aZh,
                  const _Float16* __restrict__ Bm, int ldb, long bZb, long bZh,
                  int M, int N, int K, int Kvld, int nhDiv,
                  const float* __restrict__ bias,
                  const float* __restrict__ res, int ldr, long rZb, long rZh,
                  float alpha, int act, int out16,
                  float* __restrict__ Cf, _Float16* __restrict__ Ch,
                  int ldc, long cZb, long cZh)
{
    const int z  = blockIdx.z;
    const int zb = z / nhDiv;
    const int zh = z - zb * nhDiv;
    const _Float16* Ap = A  + (long)zb * aZb + (long)zh * aZh;
    const _Float16* Bp = Bm + (long)zb * bZb + (long)zh * bZh;

    const int lane  = threadIdx.x & 31;
    const int wave  = threadIdx.x >> 5;
    const int mBase = blockIdx.y * 64 + (wave >> 1) * 16;
    const int nBase = blockIdx.x * 64 + (wave & 1) * 32;

    v8f acc0 = {};
    v8f acc1 = {};

    // A: M = lane%16 (clamped), K split 0/8 by lane-half per ISA layout.
    const int mRow = min(mBase + (lane & 15), M - 1);
    const int kb   = (lane >> 4) << 3;     // 0 or 8
    const _Float16* aRow = Ap + (size_t)mRow * lda + kb;
    const int kRow = lane;                 // B fragment row (K within tile)

    // main-loop bound: largest multiple of 32 with k0 + 31 < Kvld (Kvld >= 32)
    const int kEnd = min(K, (((Kvld - 32) >> 5) << 5) + 32);

    const _Float16* bRow = Bp + (size_t)kb * ldb + nBase + (lane & 15);
    const size_t bStep = (size_t)32 * ldb;

    int k0 = 0;
    for (; k0 < kEnd; k0 += 32) {
        // ---- A fragment: two 16B loads ----
        v8h lo = *(const v8h*)(aRow + k0);
        v8h hi = *(const v8h*)(aRow + k0 + 16);
        v16h a;
        #pragma unroll
        for (int j = 0; j < 8; ++j) { a[j] = lo[j]; a[8 + j] = hi[j]; }

        // ---- B fragments: N = lane%16, K = kb+j / 16+kb+j (same as A), ptr-increment ----
        v16h b0, b1;
        #pragma unroll
        for (int j = 0; j < 8; ++j) {
            b0[j] = bRow[(size_t)j * ldb]; b0[8 + j] = bRow[(size_t)(16 + j) * ldb];
            b1[j] = bRow[(size_t)j * ldb + 16]; b1[8 + j] = bRow[(size_t)(16 + j) * ldb + 16];
        }
        bRow += bStep;

        acc0 = __builtin_amdgcn_wmma_f32_16x16x32_f16(false, a, false, b0,
                                                      (short)0, acc0, false, false);
        acc1 = __builtin_amdgcn_wmma_f32_16x16x32_f16(false, a, false, b1,
                                                      (short)0, acc1, false, false);
    }
    // tail: only reached for K-padded att@v (k rows clamped; weights are 0)
    for (; k0 < K; k0 += 32) {
        v8h lo = *(const v8h*)(aRow + k0);
        v8h hi = *(const v8h*)(aRow + k0 + 16);
        v16h a;
        #pragma unroll
        for (int j = 0; j < 8; ++j) { a[j] = lo[j]; a[8 + j] = hi[j]; }

        const _Float16* bc = Bp + nBase + (lane & 15);
        v16h b0, b1;
        #pragma unroll
        for (int j = 0; j < 8; ++j) {
            const int ka = min(k0 + kb + j, Kvld - 1);
            const int kc = min(k0 + kb + 16 + j, Kvld - 1);
            b0[j] = bc[(size_t)ka * ldb]; b0[8 + j] = bc[(size_t)kc * ldb];
            b1[j] = bc[(size_t)ka * ldb + 16]; b1[8 + j] = bc[(size_t)kc * ldb + 16];
        }

        acc0 = __builtin_amdgcn_wmma_f32_16x16x32_f16(false, a, false, b0,
                                                      (short)0, acc0, false, false);
        acc1 = __builtin_amdgcn_wmma_f32_16x16x32_f16(false, a, false, b1,
                                                      (short)0, acc1, false, false);
    }

    // ---- epilogue: C layout lane n = lane%16, rows r + 8*(lane/16) ----
    const long coff = (long)zb * cZb + (long)zh * cZh;
    const float* rp = res ? res + (long)zb * rZb + (long)zh * rZh : nullptr;
    const int nCol = lane & 15;
    const int rOff = (lane >> 4) << 3;
    #pragma unroll
    for (int r = 0; r < 8; ++r) {
        int m = mBase + rOff + r;
        if (m >= M) continue;
        #pragma unroll
        for (int t = 0; t < 2; ++t) {
            int n = nBase + (t ? 16 : 0) + nCol;
            if (n >= N) continue;
            float v = (t ? acc1[r] : acc0[r]) * alpha;
            if (bias) v += bias[n];
            if (act == 1) v = 0.5f * v * (1.f + erff(v * 0.70710678118654752f));
            if (rp) v += rp[(size_t)m * ldr + n];
            if (out16) Ch[coff + (size_t)m * ldc + n] = (_Float16)v;
            else       Cf[coff + (size_t)m * ldc + n] = v;
        }
    }
}

// ---------------- K transpose: Kt[z][d][s] = K[b][s][h*64+d], zero-padded ----
__global__ __launch_bounds__(256)
void fp_kt(const _Float16* __restrict__ QKV, _Float16* __restrict__ Kt, int S)
{
    int z = blockIdx.x;                 // b*NH + h
    int b = z / NH, h = z - b * NH;
    const _Float16* src = QKV + (long)b * SMAX * TD + Dm + h * HD;
    _Float16* dst = Kt + (long)z * HD * SPAD;
    for (int i = threadIdx.x; i < HD * SPAD; i += 256) {
        int d = i >> 8;                 // i / SPAD
        int s = i & (SPAD - 1);         // contiguous writes
        dst[i] = (s < S) ? src[(size_t)s * TD + d] : (_Float16)0.f;
    }
}

// ---------------- LayerNorm (row of 768 = 256 threads x 3) ----------------
template<bool OUT16>
__global__ __launch_bounds__(256)
void fp_ln(const float* __restrict__ X, long xBatch, int S,
           const float* __restrict__ g, const float* __restrict__ b,
           _Float16* __restrict__ oh, long ohBatch,
           float* __restrict__ of, long ofBatch)
{
    int row = blockIdx.x;
    int bz = row / S, s = row - bz * S;
    const float* x = X + (long)bz * xBatch + (size_t)s * Dm;
    int t = threadIdx.x;
    float v0 = x[t], v1 = x[t + 256], v2 = x[t + 512];
    __shared__ float red[256];
    red[t] = v0 + v1 + v2;
    __syncthreads();
    for (int o = 128; o > 0; o >>= 1) { if (t < o) red[t] += red[t + o]; __syncthreads(); }
    float mean = red[0] / Dm;
    __syncthreads();
    float d0 = v0 - mean, d1 = v1 - mean, d2 = v2 - mean;
    red[t] = d0 * d0 + d1 * d1 + d2 * d2;
    __syncthreads();
    for (int o = 128; o > 0; o >>= 1) { if (t < o) red[t] += red[t + o]; __syncthreads(); }
    float rstd = rsqrtf(red[0] / Dm + 1e-6f);
    float y0 = d0 * rstd * g[t]       + b[t];
    float y1 = d1 * rstd * g[t + 256] + b[t + 256];
    float y2 = d2 * rstd * g[t + 512] + b[t + 512];
    if (OUT16) {
        _Float16* o16 = oh + (long)bz * ohBatch + (size_t)s * Dm;
        o16[t] = (_Float16)y0; o16[t + 256] = (_Float16)y1; o16[t + 512] = (_Float16)y2;
    } else {
        float* o32 = of + (long)bz * ofBatch + (size_t)s * Dm;
        o32[t] = y0; o32[t + 256] = y1; o32[t + 512] = y2;
    }
}

// ---------------- softmax: row of S, writes f16 with zero tail to SPAD ------
__global__ __launch_bounds__(256)
void fp_softmax(const float* __restrict__ att, _Float16* __restrict__ out, int S)
{
    long row = blockIdx.x;              // z*S + sq
    long z = row / S; int sq = (int)(row - z * S);
    const float* x = att + (z * SMAX + sq) * (long)SPAD;
    _Float16*    y = out + (z * SMAX + sq) * (long)SPAD;
    int t = threadIdx.x;
    float v = (t < S) ? x[t] : -3.4e38f;
    __shared__ float red[256];
    red[t] = v;
    __syncthreads();
    for (int o = 128; o > 0; o >>= 1) { if (t < o) red[t] = fmaxf(red[t], red[t + o]); __syncthreads(); }
    float mx = red[0];
    __syncthreads();
    float e = (t < S) ? __expf(v - mx) : 0.f;
    red[t] = e;
    __syncthreads();
    for (int o = 128; o > 0; o >>= 1) { if (t < o) red[t] += red[t + o]; __syncthreads(); }
    y[t] = (_Float16)(e / red[0]);      // tail lanes write 0
}

// ---------------- small elementwise kernels ----------------
__global__ void fp_cvt(const float* __restrict__ in, _Float16* __restrict__ out, long n)
{
    long i = (long)blockIdx.x * blockDim.x + threadIdx.x;
    if (i < n) out[i] = (_Float16)in[i];
}

// patch_w (O=768, K=768 flattened) -> f16 transposed to [K][O]
__global__ void fp_cvtT(const float* __restrict__ in, _Float16* __restrict__ out)
{
    long i = (long)blockIdx.x * blockDim.x + threadIdx.x;
    if (i < (long)Dm * Dm) {
        int k = (int)(i % Dm);
        int o = (int)(i / Dm);
        out[(long)k * Dm + o] = (_Float16)in[i];
    }
}

__global__ void fp_im2col(const float* __restrict__ inp, _Float16* __restrict__ out)
{
    long i = (long)blockIdx.x * blockDim.x + threadIdx.x;
    const long total = (long)Bb * NP * Dm;
    if (i >= total) return;
    int f = (int)(i % Dm); long t = i / Dm;
    int p = (int)(t % NP); int b = (int)(t / NP);
    int c = f >> 8, rr = (f >> 4) & 15, cc = f & 15;
    int py = p / 14, px = p % 14;
    out[i] = (_Float16)inp[((long)(b * 3 + c) * IMG + (py * 16 + rr)) * IMG + (px * 16 + cc)];
}

__global__ void fp_clspos(const float* __restrict__ cls, const float* __restrict__ pos,
                          float* __restrict__ X)
{
    int b = blockIdx.x;
    for (int d = threadIdx.x; d < Dm; d += blockDim.x)
        X[(long)b * SMAX * Dm + d] = cls[d] + pos[d];
}

__global__ void fp_prompt(const float* __restrict__ prompts, const int* __restrict__ eid,
                          const float* __restrict__ pos, float* __restrict__ X, int layer)
{
    int b = blockIdx.x;
    int e = eid[b];
    const float* src = prompts + ((long)(layer * NE + e) * LP) * Dm;
    float* dst = X + (long)b * SMAX * Dm + (long)S0 * Dm;
    for (int i = threadIdx.x; i < LP * Dm; i += blockDim.x)
        dst[i] = src[i] + pos[i % Dm];
}

__global__ __launch_bounds__(128)
void fp_head(const float* __restrict__ feat, const int* __restrict__ eid,
             const float* __restrict__ hw, const float* __restrict__ hb,
             float* __restrict__ out)
{
    int b = blockIdx.x, c = threadIdx.x;
    if (c >= NC) return;
    int e = eid[b];
    const float* f = feat + (long)b * Dm;
    const float* w = hw + (long)e * Dm * NC + c;
    float s = 0.f;
    for (int d = 0; d < Dm; ++d) s += f[d] * w[(long)d * NC];
    out[b * NC + c] = s + hb[e * NC + c];
}

// ---------------- host-side GEMM launcher ----------------
static void gemm(hipStream_t st,
                 const _Float16* A, int lda, long aZb, long aZh,
                 const _Float16* Bm, int ldb, long bZb, long bZh,
                 int M, int N, int K, int Kvld, int batches, int nhDiv,
                 const float* bias,
                 const float* res, int ldr, long rZb, long rZh,
                 float alpha, int act,
                 float* Cf, _Float16* Ch, int out16, int ldc, long cZb, long cZh)
{
    dim3 g((N + 63) / 64, (M + 63) / 64, batches), blk(256);
    fp_wmma_gemm<<<g, blk, 0, st>>>(A, lda, aZb, aZh, Bm, ldb, bZb, bZh,
        M, N, K, Kvld, nhDiv, bias, res, ldr, rZb, rZh, alpha, act, out16,
        Cf, Ch, ldc, cZb, cZh);
}

extern "C" void kernel_launch(void* const* d_in, const int* in_sizes, int n_in,
                              void* d_out, int out_size, void* d_ws, size_t ws_size,
                              hipStream_t stream)
{
    (void)in_sizes; (void)n_in; (void)out_size; (void)ws_size;
    // inputs (setup_inputs order)
    const float* inputs   = (const float*)d_in[0];
    const int*   eid      = (const int*)  d_in[1];
    const float* patch_w  = (const float*)d_in[2];
    const float* patch_b  = (const float*)d_in[3];
    const float* cls_tok  = (const float*)d_in[4];
    const float* pos      = (const float*)d_in[5];
    const float* ln1_s    = (const float*)d_in[6];
    const float* ln1_b    = (const float*)d_in[7];
    const float* qkv_w    = (const float*)d_in[8];
    const float* qkv_b    = (const float*)d_in[9];
    const float* proj_w   = (const float*)d_in[10];
    const float* proj_b   = (const float*)d_in[11];
    const float* ln2_s    = (const float*)d_in[12];
    const float* ln2_b    = (const float*)d_in[13];
    const float* fc1_w    = (const float*)d_in[14];
    const float* fc1_b    = (const float*)d_in[15];
    const float* fc2_w    = (const float*)d_in[16];
    const float* fc2_b    = (const float*)d_in[17];
    const float* norm_s   = (const float*)d_in[18];
    const float* norm_b   = (const float*)d_in[19];
    const float* prompts  = (const float*)d_in[20];
    const float* head_w   = (const float*)d_in[21];
    const float* head_b   = (const float*)d_in[22];
    float* out = (float*)d_out;

    // workspace carve (256B aligned)
    size_t off = 0;
    char* base = (char*)d_ws;
    auto carve = [&](size_t bytes) -> void* {
        void* p = base + off;
        off += (bytes + 255) & ~(size_t)255;
        return p;
    };
    const long XE   = (long)Bb * SMAX * Dm;
    const long QKVE = (long)Bb * SMAX * TD;
    const long ATTE = (long)Bb * NH * SMAX * SPAD;
    const long KTE  = (long)Bb * NH * HD * SPAD;
    const long HIDE = (long)Bb * SMAX * MLP;
    const long IME  = (long)Bb * NP * Dm;
    float*    X    = (float*)   carve(XE * 4);
    _Float16* H    = (_Float16*)carve(XE * 2);
    _Float16* QKV  = (_Float16*)carve(QKVE * 2);
    _Float16* KT   = (_Float16*)carve(KTE * 2);
    float*    ATTF = (float*)   carve(ATTE * 4);
    _Float16* ATTH = (_Float16*)carve(ATTE * 2);
    float*    O    = (float*)   carve(XE * 4);
    _Float16* HID  = (_Float16*)carve(HIDE * 2);
    _Float16* IM   = (_Float16*)carve(IME * 2);
    _Float16* W16  = (_Float16*)carve((long)Dm * MLP * 2);
    float*    FEAT = (float*)   carve((long)Bb * Dm * 4);

    auto cvt = [&](const float* src, _Float16* dst, long n) {
        fp_cvt<<<dim3((unsigned)((n + 255) / 256)), 256, 0, stream>>>(src, dst, n);
    };

    // ---- patch embed: im2col + GEMM (bias=patch_b, residual=pos_embed rows 1..196)
    fp_cvtT<<<dim3((unsigned)(((long)Dm * Dm + 255) / 256)), 256, 0, stream>>>(patch_w, W16);
    fp_im2col<<<dim3((unsigned)((IME + 255) / 256)), 256, 0, stream>>>(inputs, IM);
    fp_clspos<<<Bb, 256, 0, stream>>>(cls_tok, pos, X);
    gemm(stream,
         IM, Dm, (long)NP * Dm, 0,
         W16, Dm, 0, 0,
         NP, Dm, Dm, Dm, Bb, 1,
         patch_b, pos + Dm, Dm, 0, 0,
         1.f, 0,
         X + Dm, nullptr, 0, Dm, (long)SMAX * Dm, 0);

    // ---- transformer layers ----
    for (int i = 0; i < Lc; ++i) {
        int S = (i < NPL) ? SMAX : S0;
        if (i < NPL)
            fp_prompt<<<Bb, 256, 0, stream>>>(prompts, eid, pos, X, i);

        // LN1 -> H (f16)
        fp_ln<true><<<Bb * S, 256, 0, stream>>>(X, (long)SMAX * Dm, S,
            ln1_s + (long)i * Dm, ln1_b + (long)i * Dm, H, (long)SMAX * Dm, nullptr, 0);

        // qkv = H @ qkv_w[i] + qkv_b[i]  -> QKV (f16)
        cvt(qkv_w + (long)i * Dm * TD, W16, (long)Dm * TD);
        gemm(stream,
             H, Dm, (long)SMAX * Dm, 0,
             W16, TD, 0, 0,
             S, TD, Dm, Dm, Bb, 1,
             qkv_b + (long)i * TD, nullptr, 0, 0, 0,
             1.f, 0,
             nullptr, QKV, 1, TD, (long)SMAX * TD, 0);

        // K^T per (b,h): Kt[z][d][s], zero-padded to SPAD
        fp_kt<<<Bb * NH, 256, 0, stream>>>(QKV, KT, S);

        // scores = (q @ k^T) * scale  -> ATTF (f32)   [fast contiguous-N path]
        gemm(stream,
             QKV, TD, (long)SMAX * TD, HD,             // A = q
             KT, SPAD, (long)NH * HD * SPAD, (long)HD * SPAD,
             S, S, HD, HD, Bb * NH, NH,
             nullptr, nullptr, 0, 0, 0,
             0.125f, 0,
             ATTF, nullptr, 0, SPAD, (long)NH * SMAX * SPAD, (long)SMAX * SPAD);

        // softmax -> ATTH (f16, zero-padded to SPAD)
        fp_softmax<<<Bb * NH * S, 256, 0, stream>>>(ATTF, ATTH, S);

        // o = att @ v  -> O (f32)  [K padded to SPAD; v rows >= S clamped, weights 0]
        gemm(stream,
             ATTH, SPAD, (long)NH * SMAX * SPAD, (long)SMAX * SPAD,
             QKV + 2 * Dm, TD, (long)SMAX * TD, HD,
             S, HD, SPAD, S, Bb * NH, NH,
             nullptr, nullptr, 0, 0, 0,
             1.f, 0,
             O, nullptr, 0, Dm, (long)SMAX * Dm, HD);

        // x = x + o @ proj_w[i] + proj_b[i]
        cvt(O, H, XE);
        cvt(proj_w + (long)i * Dm * Dm, W16, (long)Dm * Dm);
        gemm(stream,
             H, Dm, (long)SMAX * Dm, 0,
             W16, Dm, 0, 0,
             S, Dm, Dm, Dm, Bb, 1,
             proj_b + (long)i * Dm, X, Dm, (long)SMAX * Dm, 0,
             1.f, 0,
             X, nullptr, 0, Dm, (long)SMAX * Dm, 0);

        // LN2 -> H
        fp_ln<true><<<Bb * S, 256, 0, stream>>>(X, (long)SMAX * Dm, S,
            ln2_s + (long)i * Dm, ln2_b + (long)i * Dm, H, (long)SMAX * Dm, nullptr, 0);

        // hid = gelu(H @ fc1_w[i] + fc1_b[i]) -> HID (f16)
        cvt(fc1_w + (long)i * Dm * MLP, W16, (long)Dm * MLP);
        gemm(stream,
             H, Dm, (long)SMAX * Dm, 0,
             W16, MLP, 0, 0,
             S, MLP, Dm, Dm, Bb, 1,
             fc1_b + (long)i * MLP, nullptr, 0, 0, 0,
             1.f, 1,
             nullptr, HID, 1, MLP, (long)SMAX * MLP, 0);

        // x = x + hid @ fc2_w[i] + fc2_b[i]
        cvt(fc2_w + (long)i * MLP * Dm, W16, (long)MLP * Dm);
        gemm(stream,
             HID, MLP, (long)SMAX * MLP, 0,
             W16, Dm, 0, 0,
             S, Dm, MLP, MLP, Bb, 1,
             fc2_b + (long)i * Dm, X, Dm, (long)SMAX * Dm, 0,
             1.f, 0,
             X, nullptr, 0, Dm, (long)SMAX * Dm, 0);
        // truncation to S0 is implicit: next layer only touches rows < S.
    }

    // ---- final LN on CLS row -> FEAT (f32), then expert head ----
    fp_ln<false><<<Bb, 256, 0, stream>>>(X, (long)SMAX * Dm, 1,
        norm_s, norm_b, nullptr, 0, FEAT, Dm);
    fp_head<<<Bb, 128, 0, stream>>>(FEAT, eid, head_w, head_b, out);
}
